// GAD_layer_80582176408328
// MI455X (gfx1250) — matrix operand order repaired and checked by reading, not compiled, not run on hardware
//
#include <hip/hip_runtime.h>
#include <hip/hip_bf16.h>

// ---------------------------------------------------------------------------
// GAD layer for MI455X (gfx1250): spectral diffusion + DGN aggregation + MLPs.
// GEMMs on v_wmma_f32_16x16x32_bf16 (fp32 accumulate); edge scatter on fp32
// global atomics. wave32, 256-thread blocks = 8 waves, 16 rows per wave.
// A fragments = two contiguous 8-float runs per lane (b128 loads); B fragments
// pre-packed bf16 in WMMA layout (two uint4 loads per tile).
// ---------------------------------------------------------------------------

typedef __attribute__((ext_vector_type(16))) __bf16 v16bf;
typedef __attribute__((ext_vector_type(8)))  float  v8f;

#define H_DIM     128
#define K_EIG     32
#define AVG_D_LOG 2.5649493574615367f   // log(13.0)
#define EPS_F     1e-8f
#define ENC_NEG_INF 0x007FFFFFu         // monotonic-uint encoding of -inf

union FragAB { v16bf v; unsigned int u[8]; uint4 q[2]; };

__device__ __forceinline__ unsigned int f2bf_pair(float a, float b) {
    unsigned int ua = __float_as_uint(a);
    ua = (ua + 0x7FFFu + ((ua >> 16) & 1u)) >> 16;      // RNE f32->bf16
    unsigned int ub = __float_as_uint(b);
    ub = (ub + 0x7FFFu + ((ub >> 16) & 1u)) >> 16;
    return (ua & 0xFFFFu) | (ub << 16);
}

// Load 8 contiguous floats (32B aligned), scale, pack into 4 bf16-pair dwords.
__device__ __forceinline__ void pack_run8(const float* __restrict__ src,
                                          float scale, unsigned int* u) {
    float4 p0 = ((const float4*)src)[0];
    float4 p1 = ((const float4*)src)[1];
    u[0] = f2bf_pair(p0.x * scale, p0.y * scale);
    u[1] = f2bf_pair(p0.z * scale, p0.w * scale);
    u[2] = f2bf_pair(p1.x * scale, p1.y * scale);
    u[3] = f2bf_pair(p1.z * scale, p1.w * scale);
}

// ---------------------------------------------------------------------------
__global__ void k_maxinit(unsigned int* __restrict__ maxb, long long total) {
    long long i = (long long)blockIdx.x * 256 + threadIdx.x;
    if (i < total) maxb[i] = ENC_NEG_INF;
}

// coeff[k][h] += sum_n eigvec[n][k] * deg[n] * x[n][h]; register partials +
// one atomicAdd round per block (4096 targets, negligible contention).
__global__ void k_coeff_accum(const float* __restrict__ xfts,
                              const float* __restrict__ deg,
                              const float* __restrict__ eigvec,
                              float* __restrict__ coeff, int Nn) {
    __shared__ float ev[K_EIG];
    int tid   = threadIdx.x;
    int h     = tid & (H_DIM - 1);
    int khalf = tid >> 7;                    // 0 or 1 -> k in [khalf*16, +16)
    float acc[16];
#pragma unroll
    for (int j = 0; j < 16; ++j) acc[j] = 0.f;
    int n0 = blockIdx.x * 128;
    int n1 = min(n0 + 128, Nn);
    for (int n = n0; n < n1; ++n) {
        if (tid < K_EIG) ev[tid] = eigvec[(long long)n * K_EIG + tid];
        __syncthreads();
        float xw = xfts[(long long)n * H_DIM + h] * deg[n];
#pragma unroll
        for (int j = 0; j < 16; ++j) acc[j] += ev[khalf * 16 + j] * xw;
        __syncthreads();
    }
#pragma unroll
    for (int j = 0; j < 16; ++j)
        atomicAdd(&coeff[(khalf * 16 + j) * H_DIM + h], acc[j]);
}

// scale coeff by exp(-lambda_k * softplus(t_h)) and pack into bf16 B-fragment
// layout: u32 index = (ct*32 + lane)*8 + v ; lane half h holds K 8h..8h+7 and 16+8h..16+8h+7.
__global__ void k_coeff_pack(const float* __restrict__ coeff,
                             const float* __restrict__ eigval,
                             const float* __restrict__ dtime,
                             unsigned int* __restrict__ cpk) {
    int idx  = blockIdx.x * 256 + threadIdx.x;     // < 2048
    int v    = idx & 7;
    int lane = (idx >> 3) & 31;
    int n    = ((idx >> 8) & 7) * 16 + (lane & 15);
    int k0   = (v >> 2) * 16 + (lane >> 4) * 8 + 2 * (v & 3);
    float t  = dtime[n];
    t = (t > 20.f) ? t : log1pf(expf(t));          // softplus
    float s0 = coeff[k0 * H_DIM + n] * expf(-eigval[k0] * t);
    float s1 = coeff[(k0 + 1) * H_DIM + n] * expf(-eigval[k0 + 1] * t);
    cpk[idx] = f2bf_pair(s0, s1);
}

// pack row-major fp32 weight [ktiles*32][128] into bf16 WMMA B fragments.
__global__ void k_pack_w(const float* __restrict__ W,
                         unsigned int* __restrict__ wpk, int ktiles) {
    int idx = blockIdx.x * 256 + threadIdx.x;
    if (idx >= ktiles * 2048) return;
    int v    = idx & 7;
    int lane = (idx >> 3) & 31;
    int ct   = (idx >> 8) & 7;
    int kt   = idx >> 11;
    int n    = ct * 16 + (lane & 15);
    int k    = kt * 32 + (v >> 2) * 16 + (lane >> 4) * 8 + 2 * (v & 3);
    wpk[idx] = f2bf_pair(W[k * H_DIM + n], W[(k + 1) * H_DIM + n]);
}

// diffusion_out = eigvec[N,32] @ coeff[32,128] : one bf16 WMMA k-step.
__global__ void k_diffusion(const float* __restrict__ eigvec,
                            const unsigned int* __restrict__ cpk,
                            float* __restrict__ xout, int Nn) {
    int lane    = threadIdx.x & 31;
    int wave    = threadIdx.x >> 5;
    int rowbase = blockIdx.x * 128 + wave * 16;
    int half    = lane >> 4;
    int r       = rowbase + (lane & 15);
    bool rv     = (r < Nn);
    int rc      = rv ? r : 0;
    float as    = rv ? 1.f : 0.f;
    FragAB a;
    pack_run8(eigvec + (long long)rc * K_EIG + half * 8,      as, a.u);
    pack_run8(eigvec + (long long)rc * K_EIG + 16 + half * 8, as, a.u + 4);
    bool fulltile = (rowbase + 16) <= Nn;
#pragma unroll
    for (int ct = 0; ct < 8; ++ct) {
        FragAB b;
        const uint4* bp = (const uint4*)&cpk[(ct * 32 + lane) * 8];
        b.q[0] = bp[0]; b.q[1] = bp[1];
        v8f c = {};
        c = __builtin_amdgcn_wmma_f32_16x16x32_bf16(false, a.v, false, b.v,
                                                    (short)0, c, false, false);
        int col = ct * 16 + (lane & 15);
        long long o = (long long)(rowbase + half * 8) * H_DIM + col;
        if (fulltile) {
#pragma unroll
            for (int rr = 0; rr < 8; ++rr) xout[o + (long long)rr * H_DIM] = c[rr];
        } else {
#pragma unroll
            for (int rr = 0; rr < 8; ++rr)
                if (rowbase + half * 8 + rr < Nn) xout[o + (long long)rr * H_DIM] = c[rr];
        }
    }
}

// edge scatter: sum, float-max (monotonic u32), |w|-weighted sum, F-weighted sum.
__global__ void k_edge(const int* __restrict__ ei, const float* __restrict__ Fn,
                       const float* __restrict__ x, float* __restrict__ sum,
                       unsigned int* __restrict__ maxb, float* __restrict__ wsum,
                       float* __restrict__ fsum, float* __restrict__ sumw, int Ee) {
    int tid = threadIdx.x;
    long long e = (long long)blockIdx.x * 2 + (tid >> 7);
    if (e >= Ee) return;
    int h   = tid & 127;
    int s   = ei[e];
    int d   = ei[Ee + e];
    float F = Fn[e];
    float w = fabsf(F);
    float m = x[(long long)s * H_DIM + h];
    long long o = (long long)d * H_DIM + h;
    atomicAdd(&sum[o], m);
    unsigned int mb = __float_as_uint(m);
    mb = (mb & 0x80000000u) ? ~mb : (mb | 0x80000000u);
    atomicMax(&maxb[o], mb);
    atomicAdd(&wsum[o], w * m);
    atomicAdd(&fsum[o], F * m);
    if (h == 0) atomicAdd(&sumw[d], w);
}

__global__ void k_finalize(float* __restrict__ sum, unsigned int* __restrict__ maxb,
                           float* __restrict__ wsum, float* __restrict__ fsum,
                           const float* __restrict__ sumw, const float* __restrict__ deg,
                           const float* __restrict__ Fdig, const float* __restrict__ x,
                           int Nn) {
    long long idx = (long long)blockIdx.x * 256 + threadIdx.x;
    if (idx >= (long long)Nn * H_DIM) return;
    int n = (int)(idx >> 7);
    sum[idx] = sum[idx] / deg[n];                              // mean_agg
    unsigned int u = maxb[idx];
    float mx = (u == ENC_NEG_INF) ? 0.f
             : ((u & 0x80000000u) ? __uint_as_float(u & 0x7FFFFFFFu)
                                  : __uint_as_float(~u));
    ((float*)maxb)[idx] = mx;                                  // max_agg
    wsum[idx] = wsum[idx] / (sumw[n] + EPS_F);                 // dir_av
    fsum[idx] = fsum[idx] - Fdig[n] * x[idx];                  // dir_dx
}

// feats column block -> (source pointer, scale). Runs of 8 never cross blocks.
__device__ __forceinline__ const float* feat_run(int c, long long rb,
                                                 const float* x, const float* a0,
                                                 const float* a1, const float* a2,
                                                 const float* a3, float amp,
                                                 float* scale) {
    int b = c >> 7;
    int h = c & 127;
    *scale = (b >= 5) ? amp : 1.f;
    const float* p;
    switch (b) {
        case 0: p = x;  break;
        case 1: case 5: p = a0; break;
        case 2: case 6: p = a1; break;
        case 3: case 7: p = a2; break;
        default:        p = a3; break;
    }
    return p + rb + h;
}

// h = relu(feats[N,9H] @ W_post + b_post) * norm_n ; feats built on the fly.
__global__ void k_gemm_post(const float* __restrict__ x, const float* __restrict__ mean,
                            const float* __restrict__ maxf, const float* __restrict__ dav,
                            const float* __restrict__ ddx, const float* __restrict__ deg,
                            const unsigned int* __restrict__ wpk,
                            const float* __restrict__ bpost,
                            const float* __restrict__ normn,
                            float* __restrict__ dgn, int Nn) {
    int lane    = threadIdx.x & 31;
    int wave    = threadIdx.x >> 5;
    int rowbase = blockIdx.x * 128 + wave * 16;
    int half    = lane >> 4;
    int r       = rowbase + (lane & 15);
    bool rv     = (r < Nn);
    int rc      = rv ? r : 0;
    long long rb = (long long)rc * H_DIM;
    float amp   = log1pf(deg[rc]) / AVG_D_LOG;
    v8f acc[8] = {};
    for (int kt = 0; kt < (9 * H_DIM) / 32; ++kt) {            // 36 k-steps
        FragAB a;
        float s0, s1;
        int c0 = kt * 32 + half * 8;
        const float* p0 = feat_run(c0,      rb, x, mean, maxf, dav, ddx, amp, &s0);
        const float* p1 = feat_run(c0 + 16, rb, x, mean, maxf, dav, ddx, amp, &s1);
        pack_run8(p0, rv ? s0 : 0.f, a.u);
        pack_run8(p1, rv ? s1 : 0.f, a.u + 4);
        if (kt + 1 < 36) __builtin_prefetch(&wpk[((kt + 1) * 8) * 256], 0, 0);
#pragma unroll
        for (int ct = 0; ct < 8; ++ct) {
            FragAB b;
            const uint4* bp = (const uint4*)&wpk[((kt * 8 + ct) * 32 + lane) * 8];
            b.q[0] = bp[0]; b.q[1] = bp[1];
            acc[ct] = __builtin_amdgcn_wmma_f32_16x16x32_bf16(
                false, a.v, false, b.v, (short)0, acc[ct], false, false);
        }
    }
    bool fulltile = (rowbase + 16) <= Nn;
#pragma unroll
    for (int ct = 0; ct < 8; ++ct) {
        int col  = ct * 16 + (lane & 15);
        float bb = bpost[col];
        int row0 = rowbase + half * 8;
        long long o = (long long)row0 * H_DIM + col;
        if (fulltile) {
#pragma unroll
            for (int rr = 0; rr < 8; ++rr) {
                float v = acc[ct][rr] + bb;
                v = v > 0.f ? v : 0.f;
                dgn[o + (long long)rr * H_DIM] = v * normn[row0 + rr];
            }
        } else {
#pragma unroll
            for (int rr = 0; rr < 8; ++rr) {
                if (row0 + rr < Nn) {
                    float v = acc[ct][rr] + bb;
                    v = v > 0.f ? v : 0.f;
                    dgn[o + (long long)rr * H_DIM] = v * normn[row0 + rr];
                }
            }
        }
    }
}

// out = node_fts + [x | dgn] @ W_last + b_last ; K = 2H = 256 -> 8 k-tiles.
__global__ void k_gemm_last(const float* __restrict__ x, const float* __restrict__ dgn,
                            const unsigned int* __restrict__ wpk,
                            const float* __restrict__ blast,
                            const float* __restrict__ nfts,
                            float* __restrict__ out, int Nn) {
    int lane    = threadIdx.x & 31;
    int wave    = threadIdx.x >> 5;
    int rowbase = blockIdx.x * 128 + wave * 16;
    int half    = lane >> 4;
    int r       = rowbase + (lane & 15);
    bool rv     = (r < Nn);
    long long rb = (long long)(rv ? r : 0) * H_DIM;
    float as    = rv ? 1.f : 0.f;
    v8f acc[8] = {};
#pragma unroll
    for (int kt = 0; kt < 8; ++kt) {                           // K = 256
        FragAB a;
        int c0 = kt * 32 + half * 8;
        int c1 = c0 + 16;
        const float* p0 = (c0 < H_DIM) ? (x + rb + c0) : (dgn + rb + c0 - H_DIM);
        const float* p1 = (c1 < H_DIM) ? (x + rb + c1) : (dgn + rb + c1 - H_DIM);
        pack_run8(p0, as, a.u);
        pack_run8(p1, as, a.u + 4);
#pragma unroll
        for (int ct = 0; ct < 8; ++ct) {
            FragAB b;
            const uint4* bp = (const uint4*)&wpk[((kt * 8 + ct) * 32 + lane) * 8];
            b.q[0] = bp[0]; b.q[1] = bp[1];
            acc[ct] = __builtin_amdgcn_wmma_f32_16x16x32_bf16(
                false, a.v, false, b.v, (short)0, acc[ct], false, false);
        }
    }
    bool fulltile = (rowbase + 16) <= Nn;
#pragma unroll
    for (int ct = 0; ct < 8; ++ct) {
        int col  = ct * 16 + (lane & 15);
        float bb = blast[col];
        long long o = (long long)(rowbase + half * 8) * H_DIM + col;
        if (fulltile) {
#pragma unroll
            for (int rr = 0; rr < 8; ++rr) {
                long long oo = o + (long long)rr * H_DIM;
                out[oo] = nfts[oo] + acc[ct][rr] + bb;
            }
        } else {
#pragma unroll
            for (int rr = 0; rr < 8; ++rr) {
                if (rowbase + half * 8 + rr < Nn) {
                    long long oo = o + (long long)rr * H_DIM;
                    out[oo] = nfts[oo] + acc[ct][rr] + bb;
                }
            }
        }
    }
}

// ---------------------------------------------------------------------------
extern "C" void kernel_launch(void* const* d_in, const int* in_sizes, int n_in,
                              void* d_out, int out_size, void* d_ws, size_t ws_size,
                              hipStream_t stream) {
    const float* node_fts = (const float*)d_in[0];
    const int*   ei       = (const int*)d_in[2];
    const float* Fn       = (const float*)d_in[3];
    const float* Fdig     = (const float*)d_in[4];
    const float* deg      = (const float*)d_in[5];
    const float* eigval   = (const float*)d_in[8];
    const float* eigvec   = (const float*)d_in[9];
    const float* normn    = (const float*)d_in[11];
    const float* dtime    = (const float*)d_in[13];
    const float* Wpost    = (const float*)d_in[14];
    const float* bpost    = (const float*)d_in[15];
    const float* Wlast    = (const float*)d_in[16];
    const float* blast    = (const float*)d_in[17];

    const int Nn = in_sizes[0] / H_DIM;
    const int Ee = in_sizes[2] / 2;
    const long long NH = (long long)Nn * H_DIM;

    // workspace layout
    float* ws    = (float*)d_ws;
    float* sum   = ws;                         // [N,H] -> mean_agg in place
    float* wsum  = sum  + NH;                  // [N,H] -> dir_av in place
    float* fsum  = wsum + NH;                  // [N,H] -> dir_dx in place
    float* sumw  = fsum + NH;                  // [N]
    float* coeff = sumw + Nn;                  // [32,128]
    unsigned int* maxb = (unsigned int*)(coeff + K_EIG * H_DIM);  // [N,H]
    float* x     = (float*)(maxb + NH);        // diffusion_out [N,H]
    float* dgn   = x + NH;                     // [N,H]
    unsigned int* cpk  = (unsigned int*)(dgn + NH);  // 2048 u32
    unsigned int* wppk = cpk + 2048;                 // 36*2048 u32 (W_post)
    unsigned int* wlpk = wppk + 36 * 2048;           // 8*2048 u32 (W_last)

    hipMemsetAsync(ws, 0, (size_t)(3 * NH + Nn + K_EIG * H_DIM) * sizeof(float), stream);

    const dim3 B(256);
    const int nb128 = (Nn + 127) / 128;

    k_maxinit<<<(int)((NH + 255) / 256), B, 0, stream>>>(maxb, NH);
    k_pack_w<<<36 * 8, B, 0, stream>>>(Wpost, wppk, 36);
    k_pack_w<<<8 * 8, B, 0, stream>>>(Wlast, wlpk, 8);
    k_coeff_accum<<<nb128, B, 0, stream>>>(node_fts, deg, eigvec, coeff, Nn);
    k_coeff_pack<<<8, B, 0, stream>>>(coeff, eigval, dtime, cpk);
    k_diffusion<<<nb128, B, 0, stream>>>(eigvec, cpk, x, Nn);
    k_edge<<<(Ee + 1) / 2, B, 0, stream>>>(ei, Fn, x, sum, maxb, wsum, fsum, sumw, Ee);
    k_finalize<<<(int)((NH + 255) / 256), B, 0, stream>>>(sum, maxb, wsum, fsum,
                                                          sumw, deg, Fdig, x, Nn);
    k_gemm_post<<<nb128, B, 0, stream>>>(x, sum, (const float*)maxb, wsum, fsum,
                                         deg, wppk, bpost, normn, dgn, Nn);
    k_gemm_last<<<nb128, B, 0, stream>>>(x, dgn, wlpk, blast, node_fts,
                                         (float*)d_out, Nn);
}
